// BiMambaBlock_62259845923213
// MI455X (gfx1250) — matrix operand order repaired and checked by reading, not compiled, not run on hardware
//
#include <hip/hip_runtime.h>

// ---------------- problem constants ----------------
#define DM    768          // d_model
#define DI    1536         // d_inner
#define DS_N  16           // d_state
#define DC    4            // d_conv
#define DTR   48           // dt_rank
#define BATCH 2
#define SEQ   2048
#define NR    (BATCH*SEQ)  // 4096 rows
#define XZN   (2*DI)       // 3072
#define XPN   (DTR+2*DS_N) // 80

typedef __attribute__((ext_vector_type(16))) __bf16 v16bf;
typedef __attribute__((ext_vector_type(8)))  __bf16 v8bf;
typedef __attribute__((ext_vector_type(8)))  float  v8f;
typedef int v4i_t __attribute__((vector_size(16)));   // b128 payload type

#ifndef __has_builtin
#define __has_builtin(x) 0
#endif
#if __has_builtin(__builtin_amdgcn_global_load_async_to_lds_b128)
#define HAVE_ASYNC 1
typedef __attribute__((address_space(1))) v4i_t* gb128_p;   // global b128 ptr
typedef __attribute__((address_space(3))) v4i_t* lb128_p;   // LDS b128 ptr
#else
#define HAVE_ASYNC 0
#endif

__device__ __forceinline__ void wait_async_lds() {
#if HAVE_ASYNC
#if __has_builtin(__builtin_amdgcn_s_wait_asynccnt)
  __builtin_amdgcn_s_wait_asynccnt(0);
#else
  asm volatile("s_wait_asynccnt 0x0" ::: "memory");
#endif
#endif
}

__device__ __forceinline__ unsigned short f32_to_bf16(float f) {
  unsigned int u = __float_as_uint(f);
  unsigned int r = u + 0x7FFFu + ((u >> 16) & 1u);   // round-to-nearest-even
  return (unsigned short)(r >> 16);
}

// ---------------- fp32 -> bf16 convert ----------------
__global__ void cvt_bf16_kernel(const float* __restrict__ src,
                                unsigned short* __restrict__ dst, int n) {
  int i = blockIdx.x * blockDim.x + threadIdx.x;
  if (i < n) dst[i] = f32_to_bf16(src[i]);
}

// ---------------- LayerNorm -> bf16 activations ----------------
__global__ void ln_kernel(const float* __restrict__ x,
                          const float* __restrict__ nw,
                          const float* __restrict__ nb,
                          unsigned short* __restrict__ hbf) {
  __shared__ float red[256];
  int row = blockIdx.x, tid = threadIdx.x;
  const float* xr = x + (size_t)row * DM;
  float s = 0.f;
  for (int i = tid; i < DM; i += 256) s += xr[i];
  red[tid] = s; __syncthreads();
  for (int off = 128; off; off >>= 1) { if (tid < off) red[tid] += red[tid + off]; __syncthreads(); }
  float mu = red[0] * (1.0f / DM);
  __syncthreads();
  float s2 = 0.f;
  for (int i = tid; i < DM; i += 256) { float d = xr[i] - mu; s2 += d * d; }
  red[tid] = s2; __syncthreads();
  for (int off = 128; off; off >>= 1) { if (tid < off) red[tid] += red[tid + off]; __syncthreads(); }
  float inv = rsqrtf(red[0] * (1.0f / DM) + 1e-5f);
  for (int i = tid; i < DM; i += 256)
    hbf[(size_t)row * DM + i] = f32_to_bf16((xr[i] - mu) * inv * nw[i] + nb[i]);
}

// ---------------- bf16 WMMA GEMM:  C[M,N] = A[M,K] * W[N,K]^T ----------------
// Block: 128 threads = 4 waves, 64x64 output tile; wave w -> rows [w*16,w*16+16),
// 4 accumulators across the 64-wide N tile (4 WMMAs per A fragment).
// B tile (64 rows x 32 K = 4KB) staged in LDS via global_load_async_to_lds_b128,
// double-buffered with s_wait_asynccnt + barrier (DMA of tile k+1 overlaps the
// 4-WMMA compute of tile k).
// mode 0: C = acc ; mode 1: C = resid + 0.5*acc   (out_proj residual epilogue)
__global__ void gemm_bf16_wmma(const __bf16* __restrict__ A,
                               const __bf16* __restrict__ W,
                               float* __restrict__ C,
                               const float* __restrict__ resid,
                               int M, int N, int K, int mode) {
  // LDS: 2 buffers x 64 rows x 80B (32 bf16 + 16B pad -> conflict-free frags)
  __shared__ __align__(16) unsigned char bsm[2][64 * 80];

  int wave = threadIdx.x >> 5;               // 0..3 (wave32)
  int lane = threadIdx.x & 31;
  int n0 = blockIdx.x * 64;                  // N block tile
  int m0 = blockIdx.y * 64 + wave * 16;      // per-wave M tile
  int half = lane >> 4;
  int l16  = lane & 15;
  const __bf16* arow = A + (size_t)(m0 + l16) * K;   // A lane: row M=l16

  // stage k-chunk [kk,kk+32) of rows [n0,n0+64) into bsm[buf]
  auto stage = [&](int buf, int kk) {
    for (int c = threadIdx.x; c < 256; c += 128) {   // 256 x 16B chunks
      int row = c >> 2, ch = c & 3;
      int nr = n0 + row; if (nr > N - 1) nr = N - 1; // clamp ragged N (N=80)
      const __bf16* g = W + (size_t)nr * K + kk + ch * 8;
      unsigned char* l = &bsm[buf][row * 80 + ch * 16];
#if HAVE_ASYNC
      __builtin_amdgcn_global_load_async_to_lds_b128(
          (gb128_p)(void*)g, (lb128_p)(void*)l, 0, 0);
#else
      *(v8bf*)l = *(const v8bf*)g;
#endif
    }
  };

  v8f c0 = {}, c1 = {}, c2 = {}, c3 = {};
  stage(0, 0);
  for (int ki = 0, kk = 0; kk < K; ++ki, kk += 32) {
    wait_async_lds();          // this wave's staged tile is in LDS
    __syncthreads();           // everyone's tile visible, prev reads done
    if (kk + 32 < K) stage((ki + 1) & 1, kk + 32);   // overlap next DMA
    int buf = ki & 1;

    if (kk + 64 < K) {         // gfx1250 global_prefetch_b8 for A stream
      __builtin_prefetch(arow + kk + 64, 0, 0);
    }
    v8bf alo = *(const v8bf*)(arow + kk + half * 8);        // K [h*8,h*8+8)
    v8bf ahi = *(const v8bf*)(arow + kk + 16 + half * 8);   // K +16
    v16bf a = __builtin_shufflevector(alo, ahi,
        0, 1, 2, 3, 4, 5, 6, 7, 8, 9, 10, 11, 12, 13, 14, 15);

#pragma unroll
    for (int j = 0; j < 4; ++j) {
      const unsigned char* bp = &bsm[buf][(j * 16 + l16) * 80 + half * 16];
      v8bf b0 = *(const v8bf*)bp;            // ds_load_b128
      v8bf b1 = *(const v8bf*)(bp + 32);
      v16bf b = __builtin_shufflevector(b0, b1,
          0, 1, 2, 3, 4, 5, 6, 7, 8, 9, 10, 11, 12, 13, 14, 15);
      v8f& acc = (j == 0) ? c0 : (j == 1) ? c1 : (j == 2) ? c2 : c3;
      acc = __builtin_amdgcn_wmma_f32_16x16x32_bf16(
          false, a, false, b, (short)0, acc, false, false);
    }
  }

  // epilogue: D element v -> row m0+half*8+v, col n0+j*16+l16
  int rbase = m0 + half * 8;
#pragma unroll
  for (int j = 0; j < 4; ++j) {
    int col = n0 + j * 16 + l16;
    if (col >= N) break;                     // ragged N guard (uniform per tile)
    v8f& acc = (j == 0) ? c0 : (j == 1) ? c1 : (j == 2) ? c2 : c3;
    if (mode == 0) {
#pragma unroll
      for (int v = 0; v < 8; ++v) C[(size_t)(rbase + v) * N + col] = acc[v];
    } else {
#pragma unroll
      for (int v = 0; v < 8; ++v) {
        size_t idx = (size_t)(rbase + v) * N + col;
        C[idx] = resid[idx] + 0.5f * acc[v];
      }
    }
  }
}

// ---------------- depthwise causal/anticausal conv + SiLU ----------------
__global__ void conv_silu_kernel(const float* __restrict__ xz,
                                 const float* __restrict__ conv_w,
                                 const float* __restrict__ conv_b,
                                 float* __restrict__ uf,
                                 unsigned short* __restrict__ ubf, int dir) {
  int idx = blockIdx.x * blockDim.x + threadIdx.x;   // over B*S*DI, d fastest
  if (idx >= NR * DI) return;
  int d = idx % DI;
  int t = (idx / DI) % SEQ;
  int b = idx / (DI * SEQ);
  float acc = conv_b[d];
#pragma unroll
  for (int k = 0; k < DC; ++k) {
    int tt = dir ? (t + (DC - 1) - k) : (t - (DC - 1) + k);
    if (tt >= 0 && tt < SEQ)
      acc += conv_w[d * DC + k] * xz[((size_t)b * SEQ + tt) * XZN + d];
  }
  float sv = acc / (1.f + __expf(-acc));             // SiLU
  uf[idx] = sv;
  ubf[idx] = f32_to_bf16(sv);
}

// ---------------- fused selective scan + gate ----------------
__global__ void scan_gate_kernel(const float* __restrict__ dbl,
                                 const float* __restrict__ uf,
                                 const float* __restrict__ xz,
                                 const float* __restrict__ dt_w,
                                 const float* __restrict__ dt_b,
                                 const float* __restrict__ A_log,
                                 const float* __restrict__ Dp,
                                 float* __restrict__ ysum, int dir) {
  int ch = blockIdx.x * blockDim.x + threadIdx.x;
  if (ch >= BATCH * DI) return;
  int b = ch / DI, d = ch % DI;
  float wdt[DTR];
#pragma unroll
  for (int r = 0; r < DTR; ++r) wdt[r] = dt_w[d * DTR + r];
  float Ac[DS_N];
#pragma unroll
  for (int s = 0; s < DS_N; ++s) Ac[s] = -__expf(A_log[d * DS_N + s]);
  float h[DS_N];
#pragma unroll
  for (int s = 0; s < DS_N; ++s) h[s] = 0.f;
  float bias = dt_b[d], Dd = Dp[d];

  for (int step = 0; step < SEQ; ++step) {
    int t = dir ? (SEQ - 1 - step) : step;
    size_t row = (size_t)b * SEQ + t;
    const float* dr = dbl + row * XPN;       // [dt(48) | B(16) | C(16)]
    float xdt = bias;
#pragma unroll
    for (int r = 0; r < DTR; ++r) xdt = fmaf(wdt[r], dr[r], xdt);
    float delta = (xdt > 20.f) ? xdt : __logf(1.f + __expf(xdt));   // softplus
    float u  = uf[row * DI + d];
    float du = delta * u;
    float y = 0.f;
#pragma unroll
    for (int s = 0; s < DS_N; ++s) {
      float dA = __expf(delta * Ac[s]);
      h[s] = fmaf(dA, h[s], du * dr[DTR + s]);          // dA*h + (delta*u)*B
      y = fmaf(h[s], dr[DTR + DS_N + s], y);            // y += h*C
    }
    y = fmaf(u, Dd, y);                                 // + u*D
    float z = xz[row * XZN + DI + d];
    float yg = y * (z / (1.f + __expf(-z)));            // * silu(z)
    size_t oi = row * DI + d;
    if (dir == 0) ysum[oi] = yg; else ysum[oi] += yg;
  }
}

// ---------------- launcher ----------------
extern "C" void kernel_launch(void* const* d_in, const int* in_sizes, int n_in,
                              void* d_out, int out_size, void* d_ws, size_t ws_size,
                              hipStream_t stream) {
  (void)in_sizes; (void)n_in; (void)out_size; (void)ws_size;
  const float* x      = (const float*)d_in[0];
  const float* norm_w = (const float*)d_in[1];
  const float* norm_b = (const float*)d_in[2];
  const float* in_w   = (const float*)d_in[3];
  const float* conv_w = (const float*)d_in[4];
  const float* conv_b = (const float*)d_in[5];
  const float* xprj_w = (const float*)d_in[6];
  const float* dt_w   = (const float*)d_in[7];
  const float* dt_b   = (const float*)d_in[8];
  const float* A_log  = (const float*)d_in[9];
  const float* Dp     = (const float*)d_in[10];
  const float* out_w  = (const float*)d_in[11];

  char* ws = (char*)d_ws;
  size_t off = 0;
  auto carve = [&](size_t bytes) -> char* {
    char* p = ws + off;
    off = (off + bytes + 255) & ~(size_t)255;
    return p;
  };
  unsigned short* inw_bf  = (unsigned short*)carve((size_t)XZN * DM * 2);
  unsigned short* xpw_bf  = (unsigned short*)carve((size_t)XPN * DI * 2);
  unsigned short* outw_bf = (unsigned short*)carve((size_t)DM * DI * 2);
  unsigned short* h_bf    = (unsigned short*)carve((size_t)NR * DM * 2);
  float*          xz      = (float*)carve((size_t)NR * XZN * 4);
  float*          ucv_f   = (float*)carve((size_t)NR * DI * 4);
  unsigned short* ucv_bf  = (unsigned short*)carve((size_t)NR * DI * 2);
  float*          dbl     = (float*)carve((size_t)NR * XPN * 4);
  float*          ysum    = (float*)carve((size_t)NR * DI * 4);
  unsigned short* y_bf    = (unsigned short*)carve((size_t)NR * DI * 2);

  // weights -> bf16
  {
    int n1 = XZN * DM, n2 = XPN * DI, n3 = DM * DI;
    cvt_bf16_kernel<<<(n1 + 255) / 256, 256, 0, stream>>>(in_w,   inw_bf,  n1);
    cvt_bf16_kernel<<<(n2 + 255) / 256, 256, 0, stream>>>(xprj_w, xpw_bf,  n2);
    cvt_bf16_kernel<<<(n3 + 255) / 256, 256, 0, stream>>>(out_w,  outw_bf, n3);
  }

  // LayerNorm -> h (bf16)
  ln_kernel<<<NR, 256, 0, stream>>>(x, norm_w, norm_b, h_bf);

  // in_proj (shared by both directions): xz[NR,3072] = h @ in_w^T
  gemm_bf16_wmma<<<dim3((XZN + 63) / 64, NR / 64), 128, 0, stream>>>(
      (const __bf16*)h_bf, (const __bf16*)inw_bf, xz, nullptr, NR, XZN, DM, 0);

  for (int dir = 0; dir < 2; ++dir) {
    conv_silu_kernel<<<((NR * DI) + 255) / 256, 256, 0, stream>>>(
        xz, conv_w, conv_b, ucv_f, ucv_bf, dir);
    gemm_bf16_wmma<<<dim3((XPN + 63) / 64, NR / 64), 128, 0, stream>>>(
        (const __bf16*)ucv_bf, (const __bf16*)xpw_bf, dbl, nullptr, NR, XPN, DI, 0);
    scan_gate_kernel<<<((BATCH * DI) + 255) / 256, 256, 0, stream>>>(
        dbl, ucv_f, xz, dt_w, dt_b, A_log, Dp, ysum, dir);
  }

  // shared out_proj with fused residual: out = x + 0.5 * (ysum @ out_w^T)
  cvt_bf16_kernel<<<((NR * DI) + 255) / 256, 256, 0, stream>>>(ysum, y_bf, NR * DI);
  gemm_bf16_wmma<<<dim3((DM + 63) / 64, NR / 64), 128, 0, stream>>>(
      (const __bf16*)y_bf, (const __bf16*)outw_bf, (float*)d_out, x, NR, DM, DI, 1);
}
